// GATBlock_14645838479517
// MI455X (gfx1250) — matrix operand order repaired and checked by reading, not compiled, not run on hardware
//
#include <hip/hip_runtime.h>
#include <hip/hip_bf16.h>

#define N_NODES  50000
#define IN_DIM   256
#define OUT_DIM  256
#define HEADS    8
#define CH       32          // channels per head
#define E_EDGES  800000
#define G_GRAPHS 64
#define NEG_SLOPE 0.2f
#define GN_EPS    1e-5f
#define E_TOT    (E_EDGES + N_NODES)      // with self loops
#define FLIP_NEG_INF 0x007FFFFFu          // fflip(-inf)

typedef __attribute__((ext_vector_type(16))) __bf16 v16bf;
typedef __attribute__((ext_vector_type(8)))  float  v8f;

// ---------------- helpers ----------------
__device__ __forceinline__ unsigned short f2bf(float f) {
    unsigned u = __float_as_uint(f);
    u += 0x7fffu + ((u >> 16) & 1u);     // round-to-nearest-even
    return (unsigned short)(u >> 16);
}
__device__ __forceinline__ unsigned fflip(float f) {   // order-preserving float->u32
    unsigned u = __float_as_uint(f);
    return (u & 0x80000000u) ? ~u : (u | 0x80000000u);
}
__device__ __forceinline__ float funflip(unsigned u) {
    u = (u & 0x80000000u) ? (u & 0x7fffffffu) : ~u;
    return __uint_as_float(u);
}

// ---------------- init fills ----------------
__global__ void fill_f32(float* p, float v, long long n) {
    long long i = (long long)blockIdx.x * blockDim.x + threadIdx.x;
    if (i < n) p[i] = v;
}
__global__ void fill_u32(unsigned* p, unsigned v, long long n) {
    long long i = (long long)blockIdx.x * blockDim.x + threadIdx.x;
    if (i < n) p[i] = v;
}

// ---------------- conversions ----------------
__global__ void cvt_x_bf16(const float* __restrict__ x, unsigned short* __restrict__ xb) {
    long long i = (long long)blockIdx.x * blockDim.x + threadIdx.x;
    if (i < (long long)N_NODES * IN_DIM) xb[i] = f2bf(x[i]);
}
// transpose W (k-major) -> Wt[n][k] bf16, for both W_l and W_r
__global__ void cvt_w_bf16(const float* __restrict__ wl, const float* __restrict__ wr,
                           unsigned short* __restrict__ wlt, unsigned short* __restrict__ wrt) {
    int i = blockIdx.x * blockDim.x + threadIdx.x;       // 0 .. 256*256-1
    if (i >= IN_DIM * OUT_DIM) return;
    int n = i >> 8, k = i & 255;
    wlt[n * IN_DIM + k] = f2bf(wl[k * OUT_DIM + n]);
    wrt[n * IN_DIM + k] = f2bf(wr[k * OUT_DIM + n]);
}

// ---------------- WMMA dual GEMM: xl = x@Wl+bl, xr = x@Wr+br ----------------
union AFrag { unsigned u[8]; v16bf v; };

__global__ __launch_bounds__(128)
void gemm_wmma(const unsigned short* __restrict__ xb,
               const unsigned short* __restrict__ wlt,
               const unsigned short* __restrict__ wrt,
               const float* __restrict__ bl, const float* __restrict__ br,
               float* __restrict__ xl, float* __restrict__ xr) {
    int wave = blockIdx.x * (blockDim.x >> 5) + (threadIdx.x >> 5);
    int lane = threadIdx.x & 31;
    int m0 = wave * 16;
    if (m0 >= N_NODES) return;
    int mlane = lane & 15;
    int hi = lane >> 4;

    // A fragment (16x32 bf16): lane holds row m0+mlane; K pairs at kt*32 + hi*8 (+16 for VGPR 4..7)
    const unsigned* xrow = (const unsigned*)(xb + (long long)(m0 + mlane) * IN_DIM);
    AFrag a[8];
#pragma unroll
    for (int kt = 0; kt < 8; ++kt) {
        int kb = kt * 16 + hi * 4;            // dword index of element kt*32 + hi*8
#pragma unroll
        for (int j = 0; j < 4; ++j) a[kt].u[j] = xrow[kb + j];
#pragma unroll
        for (int j = 0; j < 4; ++j) a[kt].u[4 + j] = xrow[kb + 8 + j];
    }

    for (int nt = 0; nt < 16; ++nt) {
        int n = nt * 16 + mlane;
        const unsigned* wlc = (const unsigned*)(wlt + (long long)n * IN_DIM);
        const unsigned* wrc = (const unsigned*)(wrt + (long long)n * IN_DIM);
        v8f accl = {};
        v8f accr = {};
#pragma unroll
        for (int kt = 0; kt < 8; ++kt) {
            // B fragment (32x16 bf16): lane holds col n; K pairs at kt*32 + hi*8 (+16 for VGPR 4..7)
            int kb = kt * 16 + hi * 4;        // dword index
            AFrag fb, fr;
#pragma unroll
            for (int j = 0; j < 4; ++j) { fb.u[j] = wlc[kb + j]; fr.u[j] = wrc[kb + j]; fb.u[4 + j] = wlc[kb + 8 + j]; fr.u[4 + j] = wrc[kb + 8 + j]; }
            accl = __builtin_amdgcn_wmma_f32_16x16x32_bf16(false, a[kt].v, false, fb.v,
                                                           (short)0, accl, false, false);
            accr = __builtin_amdgcn_wmma_f32_16x16x32_bf16(false, a[kt].v, false, fr.v,
                                                           (short)0, accr, false, false);
        }
        float biasl = bl[n], biasr = br[n];
#pragma unroll
        for (int r = 0; r < 8; ++r) {
            long long orow = m0 + r + 8 * hi;   // C layout: VGPR r -> row r (+8 for hi lanes)
            xl[orow * OUT_DIM + n] = accl[r] + biasl;
            xr[orow * OUT_DIM + n] = accr[r] + biasr;
        }
    }
}

// ---------------- edge logits + segment max (wave per edge) ----------------
__global__ __launch_bounds__(256)
void edge_logits(const long long* __restrict__ ei,
                 const float* __restrict__ xl, const float* __restrict__ xr,
                 const float* __restrict__ att,
                 float* __restrict__ logits, unsigned* __restrict__ nmax) {
    int lane = threadIdx.x & 31;
    long long e = (long long)blockIdx.x * (blockDim.x >> 5) + (threadIdx.x >> 5);
    if (e >= E_TOT) return;
    long long src, dst;
    if (e < E_EDGES) { src = ei[e]; dst = ei[E_EDGES + e]; }
    else             { src = dst = e - E_EDGES; }
    const float* pl = xl + src * OUT_DIM;
    const float* pr = xr + dst * OUT_DIM;
    float acc[HEADS];
#pragma unroll
    for (int h = 0; h < HEADS; ++h) {
        int c = h * CH + lane;
        float v = pl[c] + pr[c];
        v = (v > 0.f) ? v : v * NEG_SLOPE;          // leaky relu
        acc[h] = v * att[c];                        // att is [H][C] flat
    }
#pragma unroll
    for (int off = 16; off > 0; off >>= 1)
#pragma unroll
        for (int h = 0; h < HEADS; ++h) acc[h] += __shfl_xor(acc[h], off, 32);
    if (lane < HEADS) {
        logits[e * HEADS + lane] = acc[lane];
        atomicMax(&nmax[dst * HEADS + lane], fflip(acc[lane]));
    }
}

// ---------------- alpha = exp(logit - max), segment sum denom ----------------
__global__ __launch_bounds__(256)
void edge_alpha(const long long* __restrict__ ei,
                float* __restrict__ logits,           // in: logits, out: alpha
                const unsigned* __restrict__ nmax,
                float* __restrict__ denom) {
    long long i = (long long)blockIdx.x * blockDim.x + threadIdx.x;
    if (i >= (long long)E_TOT * HEADS) return;
    long long e = i >> 3;
    int h = (int)(i & 7);
    long long dst = (e < E_EDGES) ? ei[E_EDGES + e] : (e - E_EDGES);
    float m = funflip(nmax[dst * HEADS + h]);
    float a = __expf(logits[i] - m);
    logits[i] = a;
    atomicAdd(&denom[dst * HEADS + h], a);
}

// ---------------- message scatter: h[dst] += xl[src]*alpha (wave per edge) ----
__global__ __launch_bounds__(256)
void edge_scatter(const long long* __restrict__ ei,
                  const float* __restrict__ xl,
                  const float* __restrict__ alpha,
                  const float* __restrict__ denom,
                  float* __restrict__ hbuf) {
    int lane = threadIdx.x & 31;
    long long e = (long long)blockIdx.x * (blockDim.x >> 5) + (threadIdx.x >> 5);
    if (e >= E_TOT) return;
    long long src, dst;
    if (e < E_EDGES) { src = ei[e]; dst = ei[E_EDGES + e]; }
    else             { src = dst = e - E_EDGES; }
    const float* pl = xl + src * OUT_DIM;
    float* pd = hbuf + dst * OUT_DIM;
#pragma unroll
    for (int h = 0; h < HEADS; ++h) {
        float a = alpha[e * HEADS + h] / (denom[dst * HEADS + h] + 1e-16f);
        int c = h * CH + lane;
        atomicAdd(&pd[c], pl[c] * a);
    }
}

// ---------------- h = elu(h + bias + x); graph sums -------------------------
__global__ __launch_bounds__(256)
void node_h(const float* __restrict__ x, const float* __restrict__ bias,
            const long long* __restrict__ batch,
            float* __restrict__ hbuf, float* __restrict__ gsum,
            float* __restrict__ gcnt) {
    long long i = (long long)blockIdx.x * blockDim.x + threadIdx.x;
    if (i >= (long long)N_NODES * OUT_DIM) return;
    long long n = i >> 8;
    int c = (int)(i & 255);
    long long g = batch[n];
    float v = hbuf[i] + bias[c] + x[i];
    v = (v > 0.f) ? v : (__expf(v) - 1.f);          // ELU
    hbuf[i] = v;
    atomicAdd(&gsum[g * OUT_DIM + c], v);
    if (c == 0) atomicAdd(&gcnt[g], 1.f);
}

__global__ void graph_mean(float* __restrict__ gsum, const float* __restrict__ gcnt) {
    int i = blockIdx.x * blockDim.x + threadIdx.x;
    if (i >= G_GRAPHS * OUT_DIM) return;
    float c = gcnt[i >> 8];
    gsum[i] /= (c > 1.f ? c : 1.f);
}

__global__ __launch_bounds__(256)
void node_var(const float* __restrict__ hbuf, const long long* __restrict__ batch,
              const float* __restrict__ gmean, const float* __restrict__ mscale,
              float* __restrict__ out0, float* __restrict__ gvar) {
    long long i = (long long)blockIdx.x * blockDim.x + threadIdx.x;
    if (i >= (long long)N_NODES * OUT_DIM) return;
    long long n = i >> 8;
    int c = (int)(i & 255);
    long long g = batch[n];
    float v = hbuf[i] - mscale[c] * gmean[g * OUT_DIM + c];
    out0[i] = v;
    atomicAdd(&gvar[g * OUT_DIM + c], v * v);
}

__global__ __launch_bounds__(256)
void node_final(const float* __restrict__ out0, const long long* __restrict__ batch,
                const float* __restrict__ gvar, const float* __restrict__ gcnt,
                const float* __restrict__ w, const float* __restrict__ b,
                float* __restrict__ out) {
    long long i = (long long)blockIdx.x * blockDim.x + threadIdx.x;
    if (i >= (long long)N_NODES * OUT_DIM) return;
    long long n = i >> 8;
    int c = (int)(i & 255);
    long long g = batch[n];
    float cnt = gcnt[g]; cnt = cnt > 1.f ? cnt : 1.f;
    float var = gvar[g * OUT_DIM + c] / cnt;
    out[i] = w[c] * out0[i] * __frsqrt_rn(var + GN_EPS) + b[c];
}

// ---------------- launcher ----------------
extern "C" void kernel_launch(void* const* d_in, const int* in_sizes, int n_in,
                              void* d_out, int out_size, void* d_ws, size_t ws_size,
                              hipStream_t stream) {
    const float*     x     = (const float*)d_in[0];
    const long long* ei    = (const long long*)d_in[1];
    const long long* batch = (const long long*)d_in[2];
    const float*     Wl    = (const float*)d_in[3];
    const float*     bl    = (const float*)d_in[4];
    const float*     Wr    = (const float*)d_in[5];
    const float*     br    = (const float*)d_in[6];
    const float*     att   = (const float*)d_in[7];
    const float*     bias  = (const float*)d_in[8];
    const float*     gnw   = (const float*)d_in[9];
    const float*     gnb   = (const float*)d_in[10];
    const float*     gms   = (const float*)d_in[11];
    float* out = (float*)d_out;

    char* w = (char*)d_ws;
    size_t off = 0;
    auto alloc = [&](size_t bytes) { void* p = w + off; off = (off + bytes + 255) & ~(size_t)255; return p; };
    unsigned short* xb   = (unsigned short*)alloc((size_t)N_NODES * IN_DIM * 2);
    unsigned short* wlt  = (unsigned short*)alloc((size_t)IN_DIM * OUT_DIM * 2);
    unsigned short* wrt  = (unsigned short*)alloc((size_t)IN_DIM * OUT_DIM * 2);
    float* xl            = (float*)alloc((size_t)N_NODES * OUT_DIM * 4);
    float* xr            = (float*)alloc((size_t)N_NODES * OUT_DIM * 4);   // later reused for out0
    float* logits        = (float*)alloc((size_t)E_TOT * HEADS * 4);
    unsigned* nmax       = (unsigned*)alloc((size_t)N_NODES * HEADS * 4);
    float* denom         = (float*)alloc((size_t)N_NODES * HEADS * 4);
    float* gsum          = (float*)alloc((size_t)G_GRAPHS * OUT_DIM * 4);
    float* gvar          = (float*)alloc((size_t)G_GRAPHS * OUT_DIM * 4);
    float* gcnt          = (float*)alloc((size_t)G_GRAPHS * 4);
    (void)gvar;   // contiguous with gsum; zeroed in its own fill below

    const long long NC = (long long)N_NODES * OUT_DIM;
    auto blks = [](long long n, int t) { return (unsigned)((n + t - 1) / t); };

    // init (d_out doubles as the h accumulator; everything re-zeroed each call)
    fill_u32<<<blks((long long)N_NODES * HEADS, 256), 256, 0, stream>>>(nmax, FLIP_NEG_INF, (long long)N_NODES * HEADS);
    fill_f32<<<blks((long long)N_NODES * HEADS, 256), 256, 0, stream>>>(denom, 0.f, (long long)N_NODES * HEADS);
    fill_f32<<<blks(NC, 256), 256, 0, stream>>>(out, 0.f, NC);
    fill_f32<<<blks((long long)G_GRAPHS * OUT_DIM, 256), 256, 0, stream>>>(gsum, 0.f, (long long)G_GRAPHS * OUT_DIM);
    fill_f32<<<blks((long long)G_GRAPHS * OUT_DIM, 256), 256, 0, stream>>>(gvar, 0.f, (long long)G_GRAPHS * OUT_DIM);
    fill_f32<<<blks(G_GRAPHS, 64), 64, 0, stream>>>(gcnt, 0.f, G_GRAPHS);

    cvt_x_bf16<<<blks((long long)N_NODES * IN_DIM, 256), 256, 0, stream>>>(x, xb);
    cvt_w_bf16<<<blks(IN_DIM * OUT_DIM, 256), 256, 0, stream>>>(Wl, Wr, wlt, wrt);

    // 3125 row-waves (16 rows each), 4 waves per 128-thread block
    gemm_wmma<<<(N_NODES / 16 + 3) / 4, 128, 0, stream>>>(xb, wlt, wrt, bl, br, xl, xr);

    edge_logits<<<blks(E_TOT, 8), 256, 0, stream>>>(ei, xl, xr, att, logits, nmax);
    edge_alpha<<<blks((long long)E_TOT * HEADS, 256), 256, 0, stream>>>(ei, logits, nmax, denom);
    edge_scatter<<<blks(E_TOT, 8), 256, 0, stream>>>(ei, xl, logits, denom, out);

    node_h<<<blks(NC, 256), 256, 0, stream>>>(x, bias, batch, out, gsum, gcnt);
    graph_mean<<<blks((long long)G_GRAPHS * OUT_DIM, 256), 256, 0, stream>>>(gsum, gcnt);
    node_var<<<blks(NC, 256), 256, 0, stream>>>(out, batch, gsum, gms, xr, gvar);
    node_final<<<blks(NC, 256), 256, 0, stream>>>(xr, batch, gvar, gcnt, gnw, gnb, out);
}
